// WeightedSelfAttention_78176994722096
// MI455X (gfx1250) — matrix operand order repaired and checked by reading, not compiled, not run on hardware
//
#include <hip/hip_runtime.h>
#include <hip/hip_bf16.h>
#include <math.h>

// Problem constants (from reference): B,M,D,H,F,N_layers
constexpr int CB = 8, CM = 1024, CD = 1024, CH = 16, CF = 4096, CDK = 64;
constexpr int NLAYERS = 4;

typedef __attribute__((ext_vector_type(16))) __bf16 v16bf_t;
typedef __attribute__((ext_vector_type(8)))  float  v8f_t;

union FragBF {
  v16bf_t v;
  uint4   q[2];
};

// A-matrix fragment (16x32 bf16), rows stored row-major [row][k] in LDS.
// ISA layout: lanes 0-15 hold row M=lane, K=0..7 (v0..3) and K=16..23 (v4..7);
// lanes 16-31 hold row M=lane-16, K=8..15 and K=24..31.
__device__ __forceinline__ v16bf_t frag_ld_A(const __bf16* base, int ld, int lane) {
  const int r = lane & 15, g = lane >> 4;
  const __bf16* p = base + r * ld + 8 * g;
  FragBF f;
  f.q[0] = *(const uint4*)(p);        // 8 bf16: K = 8g .. 8g+7
  f.q[1] = *(const uint4*)(p + 16);   // 8 bf16: K = 16+8g .. 16+8g+7
  return f.v;
}

// B-matrix fragment (32x16 bf16) read from K-transposed LDS storage [n][k].
// ISA layout: lane holds column N=lane&15, K=8g..8g+7 and K=16+8g..16+8g+7 (g=lane>>4).
__device__ __forceinline__ v16bf_t frag_ld_B(const __bf16* base, int ld, int lane) {
  const int n = lane & 15, g = lane >> 4;
  const __bf16* p = base + n * ld + 8 * g;
  FragBF f;
  f.q[0] = *(const uint4*)(p);
  f.q[1] = *(const uint4*)(p + 16);
  return f.v;
}

__device__ __forceinline__ v8f_t wmma_bf16(v16bf_t a, v16bf_t b, v8f_t c) {
  return __builtin_amdgcn_wmma_f32_16x16x32_bf16(false, a, false, b, (short)0, c, false, false);
}

// ---------------------------------------------------------------------------
// GEMM: Y[M,N] = act(X[M,K] @ W[K,N] + bias[N]); act: 0=none, 1=exact GELU.
// Workgroup tile 128x128, 8 waves (4x2), each wave 32x64 (2x4 WMMA tiles).
// Software-pipelined: global loads for step k+1 issued before WMMA of step k.
// ---------------------------------------------------------------------------
#define GM_BM 128
#define GM_BN 128
#define GM_BK 32
#define GM_LD 40   // padded LDS leading dim (bf16 elems); 80B rows keep 16B align

__global__ __launch_bounds__(256) void gemm_bias_act_kernel(
    const float* __restrict__ X, const float* __restrict__ W,
    const float* __restrict__ bias, float* __restrict__ Y,
    int Mrows, int K, int N, int act)
{
  __shared__ __bf16 As[GM_BM][GM_LD];  // [m][k]
  __shared__ __bf16 Bs[GM_BN][GM_LD];  // [n][k]  (W tile, K-transposed)

  const int tid  = threadIdx.x;
  const int lane = tid & 31;
  const int wave = tid >> 5;
  const int m0 = blockIdx.y * GM_BM;
  const int n0 = blockIdx.x * GM_BN;
  const int wm = (wave >> 1) * 32;   // wave row offset within tile
  const int wn = (wave & 1) * 64;    // wave col offset within tile

  // Per-thread staging registers: A 4x float4 (128x32), B 4x float4 (32x128).
  const int am = tid >> 3,  akv = tid & 7;     // A: row, float4-col (j adds 32 rows)
  const int bkk = tid >> 5, bn4 = tid & 31;    // B: k-row, float4-col (j adds 8 k)
  float4 aregs[4], bregs[4];

  auto load_tile = [&](int k0) {
#pragma unroll
    for (int j = 0; j < 4; ++j)
      aregs[j] = *(const float4*)(X + (size_t)(m0 + am + j * 32) * K + k0 + akv * 4);
#pragma unroll
    for (int j = 0; j < 4; ++j)
      bregs[j] = *(const float4*)(W + (size_t)(k0 + bkk + j * 8) * N + n0 + bn4 * 4);
  };

  v8f_t acc[2][4] = {};

  load_tile(0);
  for (int k0 = 0; k0 < K; k0 += GM_BK) {
    __syncthreads();               // all waves done reading LDS from prior step
    // ---- store staged registers (f32 -> bf16) into LDS ----
#pragma unroll
    for (int j = 0; j < 4; ++j) {
      __bf16* dst = &As[am + j * 32][akv * 4];
      dst[0] = (__bf16)aregs[j].x; dst[1] = (__bf16)aregs[j].y;
      dst[2] = (__bf16)aregs[j].z; dst[3] = (__bf16)aregs[j].w;
    }
#pragma unroll
    for (int j = 0; j < 4; ++j) {
      const int kk = bkk + j * 8;
      Bs[bn4 * 4 + 0][kk] = (__bf16)bregs[j].x;
      Bs[bn4 * 4 + 1][kk] = (__bf16)bregs[j].y;
      Bs[bn4 * 4 + 2][kk] = (__bf16)bregs[j].z;
      Bs[bn4 * 4 + 3][kk] = (__bf16)bregs[j].w;
    }
    __syncthreads();               // tile visible to all waves

    // ---- issue next tile's global loads early (latency hidden by WMMA) ----
    if (k0 + GM_BK < K) {
      load_tile(k0 + GM_BK);
      if (k0 + 2 * GM_BK < K) {    // warm L2/WGP$ two steps ahead
        const int kpf = k0 + 2 * GM_BK;
        __builtin_prefetch((const void*)(X + (size_t)(m0 + am) * K + kpf + akv * 4), 0, 1);
        __builtin_prefetch((const void*)(W + (size_t)(kpf + bkk) * N + n0 + bn4 * 4), 0, 1);
      }
    }

    // ---- 2x4 WMMA tiles per wave ----
    v16bf_t afr[2], bfr[4];
#pragma unroll
    for (int im = 0; im < 2; ++im)
      afr[im] = frag_ld_A(&As[wm + im * 16][0], GM_LD, lane);
#pragma unroll
    for (int in = 0; in < 4; ++in)
      bfr[in] = frag_ld_B(&Bs[wn + in * 16][0], GM_LD, lane);
#pragma unroll
    for (int im = 0; im < 2; ++im)
#pragma unroll
      for (int in = 0; in < 4; ++in)
        acc[im][in] = wmma_bf16(afr[im], bfr[in], acc[im][in]);
  }

  // Epilogue: C layout -> lane holds col N=lane&15, rows M = e + 8*(lane>>4).
  const int cn = lane & 15, cg = lane >> 4;
#pragma unroll
  for (int im = 0; im < 2; ++im) {
#pragma unroll
    for (int in = 0; in < 4; ++in) {
      const int col = n0 + wn + in * 16 + cn;
      const float bv = bias[col];
#pragma unroll
      for (int e = 0; e < 8; ++e) {
        const int row = m0 + wm + im * 16 + e + 8 * cg;
        float val = acc[im][in][e] + bv;
        if (act == 1)  // exact GELU
          val = 0.5f * val * (1.0f + erff(val * 0.70710678118654752f));
        Y[(size_t)row * N + col] = val;
      }
    }
  }
}

// ---------------------------------------------------------------------------
// Attention: per (qblock of 64, head, batch). Scores (64x1024) in dynamic LDS
// (bf16, ~129KB — CDNA5 WGP has 320KB LDS), softmax, then P@V via WMMA.
// ---------------------------------------------------------------------------
#define QB     64
#define AT_LDP 1032   // Ps leading dim (bf16): 2064B rows, 16B aligned
#define AT_LDQ 72     // Qs/Ks leading dim: 144B rows
#define AT_LDV 40     // Vt leading dim: 80B rows
constexpr unsigned PS_BYTES = (unsigned)QB * AT_LDP * sizeof(__bf16); // 132096

__global__ __launch_bounds__(256) void attn_kernel(
    const float* __restrict__ q, const float* __restrict__ k,
    const float* __restrict__ v, const float* __restrict__ wts,
    float* __restrict__ out)
{
  extern __shared__ __bf16 Ps[];                 // [QB][AT_LDP] logits/probs
  __shared__ __bf16 Qs[QB][AT_LDQ];              // [q][d]
  __shared__ __bf16 Ks[64][AT_LDQ];              // [key][d]
  __shared__ __bf16 Vts[CDK][AT_LDV];            // [d][key] (32-key chunk)
  __shared__ float  linv[QB];

  const int tid  = threadIdx.x;
  const int lane = tid & 31;
  const int wave = tid >> 5;
  const int qb = blockIdx.x, h = blockIdx.y, b = blockIdx.z;

  const size_t headOff = (size_t)h * CDK;
  const size_t rowQ0   = (size_t)b * CM + (size_t)qb * QB;

  // Stage Q block (64 x 64) once.
  for (int i = tid; i < QB * CDK; i += 256) {
    const int r = i >> 6, d = i & 63;
    Qs[r][d] = (__bf16)q[(rowQ0 + r) * CD + headOff + d];
  }

  const int wq = (wave >> 1) * 16;   // 16-query tile per wave pair
  const int wk = (wave & 1) * 32;    // 32-key half

  // ---- Pass 1: logits = (Q K^T) * scale + weights[b, key] ----
  const float scale = 0.125f; // 1/sqrt(64)
  for (int kb = 0; kb < CM; kb += 64) {
    for (int i = tid; i < 64 * CDK; i += 256) {
      const int r = i >> 6, d = i & 63;
      Ks[r][d] = (__bf16)k[((size_t)b * CM + kb + r) * CD + headOff + d];
    }
    __syncthreads();

    v8f_t s[2] = {};
#pragma unroll
    for (int kk = 0; kk < 2; ++kk) {            // DK=64 -> 2 WMMA K-steps
      v16bf_t a = frag_ld_A(&Qs[wq][kk * 32], AT_LDQ, lane);
#pragma unroll
      for (int t = 0; t < 2; ++t) {
        v16bf_t bf = frag_ld_B(&Ks[wk + t * 16][kk * 32], AT_LDQ, lane);
        s[t] = wmma_bf16(a, bf, s[t]);
      }
    }
    const int cg = lane >> 4;
#pragma unroll
    for (int t = 0; t < 2; ++t) {
      const int kcol = wk + t * 16 + (lane & 15);
      const float wb = wts[(size_t)b * CM + kb + kcol];
#pragma unroll
      for (int e = 0; e < 8; ++e) {
        const int qr = wq + e + 8 * cg;
        Ps[qr * AT_LDP + kb + kcol] = (__bf16)(s[t][e] * scale + wb);
      }
    }
    __syncthreads();
  }

  // ---- Softmax over each of the 64 rows (1024 logits), wave32 reductions ----
  for (int j = 0; j < 8; ++j) {
    const int r = wave * 8 + j;
    float mx = -1e30f;
    for (int c = lane; c < CM; c += 32)
      mx = fmaxf(mx, (float)Ps[r * AT_LDP + c]);
#pragma unroll
    for (int off = 16; off >= 1; off >>= 1)
      mx = fmaxf(mx, __shfl_xor(mx, off, 32));
    float sum = 0.0f;
    for (int c = lane; c < CM; c += 32) {
      const float e = __expf((float)Ps[r * AT_LDP + c] - mx);
      sum += e;
      Ps[r * AT_LDP + c] = (__bf16)e;
    }
#pragma unroll
    for (int off = 16; off >= 1; off >>= 1)
      sum += __shfl_xor(sum, off, 32);
    if (lane == 0) linv[r] = 1.0f / sum;
  }
  __syncthreads();

  // ---- Pass 2: O = P @ V ----
  const int wd = (wave & 1) * 32;    // 32-d half
  v8f_t o[2] = {};
  for (int kb2 = 0; kb2 < CM; kb2 += 32) {
    for (int i = tid; i < CDK * 32; i += 256) {   // V chunk -> transposed LDS
      const int d = i >> 5, kk = i & 31;
      Vts[d][kk] = (__bf16)v[((size_t)b * CM + kb2 + kk) * CD + headOff + d];
    }
    __syncthreads();
    v16bf_t a = frag_ld_A(Ps + wq * AT_LDP + kb2, AT_LDP, lane);
#pragma unroll
    for (int t = 0; t < 2; ++t) {
      v16bf_t bf = frag_ld_B(&Vts[wd + t * 16][0], AT_LDV, lane);
      o[t] = wmma_bf16(a, bf, o[t]);
    }
    __syncthreads();
  }

  const int cg = lane >> 4;
#pragma unroll
  for (int t = 0; t < 2; ++t) {
    const int dcol = wd + t * 16 + (lane & 15);
#pragma unroll
    for (int e = 0; e < 8; ++e) {
      const int qr = wq + e + 8 * cg;
      out[(rowQ0 + qr) * CD + headOff + dcol] = o[t][e] * linv[qr];
    }
  }
}

// ---------------------------------------------------------------------------
// Fused residual add + LayerNorm: out = LN(x + y) * g + beta. One row/block.
// ---------------------------------------------------------------------------
__global__ __launch_bounds__(256) void add_ln_kernel(
    const float* __restrict__ x, const float* __restrict__ y,
    const float* __restrict__ g, const float* __restrict__ beta,
    float* __restrict__ outp)
{
  __shared__ float red[2][8];
  const int row = blockIdx.x, tid = threadIdx.x;
  const int lane = tid & 31, wave = tid >> 5;
  const float* xr = x + (size_t)row * CD;
  const float* yr = y + (size_t)row * CD;

  float t[4], s = 0.0f, ss = 0.0f;
#pragma unroll
  for (int j = 0; j < 4; ++j) {
    const int c = tid + j * 256;
    t[j] = xr[c] + yr[c];
    s += t[j];
    ss += t[j] * t[j];
  }
#pragma unroll
  for (int off = 16; off >= 1; off >>= 1) {
    s  += __shfl_xor(s, off, 32);
    ss += __shfl_xor(ss, off, 32);
  }
  if (lane == 0) { red[0][wave] = s; red[1][wave] = ss; }
  __syncthreads();
  float S = 0.0f, SS = 0.0f;
#pragma unroll
  for (int w = 0; w < 8; ++w) { S += red[0][w]; SS += red[1][w]; }
  const float mu  = S * (1.0f / CD);
  const float var = SS * (1.0f / CD) - mu * mu;
  const float r   = rsqrtf(var + 1e-5f);
#pragma unroll
  for (int j = 0; j < 4; ++j) {
    const int c = tid + j * 256;
    outp[(size_t)row * CD + c] = (t[j] - mu) * r * g[c] + beta[c];
  }
}

// ---------------------------------------------------------------------------
// Host orchestration. Workspace (floats): xa, xb, q, k, v, o (each B*M*D)
// then h (B*M*F)  => ~320 MB total.
// ---------------------------------------------------------------------------
extern "C" void kernel_launch(void* const* d_in, const int* in_sizes, int n_in,
                              void* d_out, int out_size, void* d_ws, size_t ws_size,
                              hipStream_t stream) {
  (void)in_sizes; (void)n_in; (void)out_size; (void)ws_size;
  const float* embeds  = (const float*)d_in[0];
  const float* weights = (const float*)d_in[1];
  const float* Wq = (const float*)d_in[2];  const float* bq = (const float*)d_in[3];
  const float* Wk = (const float*)d_in[4];  const float* bk = (const float*)d_in[5];
  const float* Wv = (const float*)d_in[6];  const float* bv = (const float*)d_in[7];
  const float* Wo = (const float*)d_in[8];  const float* bo = (const float*)d_in[9];
  const float* W1 = (const float*)d_in[10]; const float* b1 = (const float*)d_in[11];
  const float* W2 = (const float*)d_in[12]; const float* b2 = (const float*)d_in[13];
  const float* g1 = (const float*)d_in[14]; const float* be1 = (const float*)d_in[15];
  const float* g2 = (const float*)d_in[16]; const float* be2 = (const float*)d_in[17];
  float* out = (float*)d_out;

  const size_t S = (size_t)CB * CM * CD;     // 8M floats
  float* ws  = (float*)d_ws;
  float* xa  = ws;
  float* xb  = xa + S;
  float* qb_ = xb + S;
  float* kb_ = qb_ + S;
  float* vb_ = kb_ + S;
  float* ob_ = vb_ + S;
  float* hb  = ob_ + S;                      // B*M*F floats

  const dim3 blk(256);
  const dim3 gridProjD(CD / GM_BN, (CB * CM) / GM_BM);   // N=1024
  const dim3 gridProjF(CF / GM_BN, (CB * CM) / GM_BM);   // N=4096
  const dim3 gridAttn(CM / QB, CH, CB);
  const int  nRows = CB * CM;

  for (int n = 0; n < NLAYERS; ++n) {
    const float* xin  = (n == 0) ? embeds : ((n & 1) ? xa : xb);
    float*       xout = (n == NLAYERS - 1) ? out : ((n & 1) ? xb : xa);
    const size_t wD = (size_t)n * CD * CD;
    const size_t wF = (size_t)n * CD * CF;
    const size_t vD = (size_t)n * CD;
    const size_t vF = (size_t)n * CF;

    // Q, K, V projections
    gemm_bias_act_kernel<<<gridProjD, blk, 0, stream>>>(xin, Wq + wD, bq + vD, qb_, nRows, CD, CD, 0);
    gemm_bias_act_kernel<<<gridProjD, blk, 0, stream>>>(xin, Wk + wD, bk + vD, kb_, nRows, CD, CD, 0);
    gemm_bias_act_kernel<<<gridProjD, blk, 0, stream>>>(xin, Wv + wD, bv + vD, vb_, nRows, CD, CD, 0);
    // Attention with per-key additive bias (weights)
    attn_kernel<<<gridAttn, blk, PS_BYTES, stream>>>(qb_, kb_, vb_, weights, ob_);
    // Output projection (reuse q buffer)
    gemm_bias_act_kernel<<<gridProjD, blk, 0, stream>>>(ob_, Wo + wD, bo + vD, qb_, nRows, CD, CD, 0);
    // x = LN(x + o)  -> kb_
    add_ln_kernel<<<nRows, blk, 0, stream>>>(xin, qb_, g1 + vD, be1 + vD, kb_);
    // FFN: h = gelu(x @ W1 + b1); y = h @ W2 + b2 (reuse v buffer)
    gemm_bias_act_kernel<<<gridProjF, blk, 0, stream>>>(kb_, W1 + wF, b1 + vF, hb, nRows, CD, CF, 1);
    gemm_bias_act_kernel<<<gridProjD, blk, 0, stream>>>(hb, W2 + wF, b2 + vD, vb_, nRows, CF, CD, 0);
    // x = LN(x + y) -> next activations (or final output)
    add_ln_kernel<<<nRows, blk, 0, stream>>>(kb_, vb_, g2 + vD, be2 + vD, xout);
  }
}
